// DilatedAttention_48679159333046
// MI455X (gfx1250) — matrix operand order repaired and checked by reading, not compiled, not run on hardware
//
#include <hip/hip_runtime.h>
#include <hip/hip_bf16.h>
#include <cstdint>

// ---------------------------------------------------------------------------
// Dilated-attention transformer block for MI455X (gfx1250, wave32, WMMA).
// All GEMMs + both attention einsums use v_wmma_f32_16x16x32_f16.
// GEMM waves compute 32x64 output tiles: 8 WMMA per 32-deep k-step against
// 2 A-fragment + 4 B-fragment loads (B reused across both M subtiles).
// ---------------------------------------------------------------------------

#define AB   2
#define AL   2048
#define AD   1024
#define AH   16
#define AHD  64
#define ASUB 1024   // L / DILATION (per-parity subsequence length)
#define AWIN 256    // WINDOW / DILATION (window in sub-index space)

typedef __attribute__((ext_vector_type(16))) _Float16 v16h;
typedef __attribute__((ext_vector_type(8)))  _Float16 v8h;
typedef __attribute__((ext_vector_type(8)))  float    v8f;

__device__ inline v8f vzero8f() {
  v8f z = {0.f, 0.f, 0.f, 0.f, 0.f, 0.f, 0.f, 0.f};
  return z;
}

__device__ inline v8f wmma_f16(v16h a, v16h b, v8f c) {
  // (neg_a, A, neg_b, B, c_mod, C, reuse_a, reuse_b)
  return __builtin_amdgcn_wmma_f32_16x16x32_f16(false, a, false, b, (short)0, c,
                                                false, false);
}

// A fragment, 16x32 f16 (M x K), row-major A with leading dim lda (elements).
// Layout: lane<16: M=lane, elems 0..15 = K k0..k0+15;
// lane>=16: M=lane-16, elems = K k0+16..k0+31.
__device__ inline v16h load_a_frag(const _Float16* __restrict__ A, int lda,
                                   int m0, int k0) {
  const int lane = threadIdx.x & 31;
  const int hlf  = lane >> 4;
  const int row  = lane & 15;
  const _Float16* p = A + (size_t)(m0 + row) * lda + k0 + hlf * 16;
  v8h lo = *(const v8h*)(p);
  v8h hi = *(const v8h*)(p + 8);
  v16h out;
#pragma unroll
  for (int i = 0; i < 8; ++i) { out[i] = lo[i]; out[i + 8] = hi[i]; }
  return out;
}

// B fragment, 32x16 f16 (K x N), sourced from an N-major array WT[N][K]
// (i.e. B[k][n] = WT[n][k]).  Layout: lane<16: N=lane, elems = K k0..k0+15;
// lane>=16: N=lane-16, elems = K k0+16..k0+31.  One 32B contiguous load.
__device__ inline v16h load_b_frag(const _Float16* __restrict__ WT, int ldb,
                                   int n0, int k0) {
  const int lane = threadIdx.x & 31;
  const int hlf  = lane >> 4;
  const int colc = lane & 15;
  const _Float16* p = WT + (size_t)(n0 + colc) * ldb + k0 + hlf * 16;
  return *(const v16h*)(p);
}

// ---------------------------------------------------------------------------
// Weight prep: fp32 W[K][N] -> f16 WT[N][K]  (tiled LDS transpose)
// ---------------------------------------------------------------------------
__global__ void w_transpose_f16(const float* __restrict__ W,
                                _Float16* __restrict__ WT, int K, int N) {
  __shared__ float tile[32][33];
  const int n0 = blockIdx.x * 32;
  const int k0 = blockIdx.y * 32;
  const int tx = threadIdx.x;   // 0..31
  const int ty = threadIdx.y;   // 0..7
#pragma unroll
  for (int i = 0; i < 32; i += 8)
    tile[ty + i][tx] = W[(size_t)(k0 + ty + i) * N + n0 + tx];
  __syncthreads();
#pragma unroll
  for (int i = 0; i < 32; i += 8)
    WT[(size_t)(n0 + ty + i) * K + k0 + tx] = (_Float16)tile[tx][ty + i];
}

// ---------------------------------------------------------------------------
// LayerNorm over D=1024, one block (256 threads) per row, f16 output.
// ---------------------------------------------------------------------------
__global__ void ln_to_f16(const float* __restrict__ x,
                          const float* __restrict__ g,
                          const float* __restrict__ b,
                          _Float16* __restrict__ out) {
  __shared__ float red_s[8];
  __shared__ float red_q[8];
  const int row  = blockIdx.x;
  const int base = threadIdx.x * 4;
  const float* xr = x + (size_t)row * AD;

  float v0 = xr[base + 0], v1 = xr[base + 1], v2 = xr[base + 2], v3 = xr[base + 3];
  float s  = v0 + v1 + v2 + v3;
  float s2 = v0 * v0 + v1 * v1 + v2 * v2 + v3 * v3;
#pragma unroll
  for (int off = 16; off >= 1; off >>= 1) {
    s  += __shfl_xor(s, off, 32);
    s2 += __shfl_xor(s2, off, 32);
  }
  const int wid = threadIdx.x >> 5;
  if ((threadIdx.x & 31) == 0) { red_s[wid] = s; red_q[wid] = s2; }
  __syncthreads();
  float ts = 0.f, tq = 0.f;
#pragma unroll
  for (int j = 0; j < 8; ++j) { ts += red_s[j]; tq += red_q[j]; }
  const float mu  = ts * (1.0f / AD);
  const float var = tq * (1.0f / AD) - mu * mu;
  const float rs  = rsqrtf(var + 1e-5f);

  _Float16* o = out + (size_t)row * AD + base;
  o[0] = (_Float16)((v0 - mu) * rs * g[base + 0] + b[base + 0]);
  o[1] = (_Float16)((v1 - mu) * rs * g[base + 1] + b[base + 1]);
  o[2] = (_Float16)((v2 - mu) * rs * g[base + 2] + b[base + 2]);
  o[3] = (_Float16)((v3 - mu) * rs * g[base + 3] + b[base + 3]);
}

// ---------------------------------------------------------------------------
// WMMA GEMM: C[M][N] = A[M][K](f16) * WT[N][K](f16)^T + bias, + epilogue.
// Block: 128 threads (4 waves).  Wave computes a 32x64 strip of C
// (2 M-subtiles x 4 N-subtiles; B fragments reused across M).
// MODE 0: QKV scatter (q scaled by 1/sqrt(HD); v transposed+parity-compacted)
// MODE 1: out32 = C + residual (fp32)
// MODE 2: out16 = gelu_exact(C) (f16)
// ---------------------------------------------------------------------------
template <int MODE>
__global__ void gemm_wmma(const _Float16* __restrict__ A,
                          const _Float16* __restrict__ WT,
                          const float* __restrict__ bias,
                          const float* __restrict__ res,
                          float* __restrict__ out32,
                          _Float16* __restrict__ out16,
                          _Float16* __restrict__ qf,
                          _Float16* __restrict__ kf,
                          _Float16* __restrict__ vtp,
                          int M, int N, int K) {
  const int lane = threadIdx.x & 31;
  const int wv   = threadIdx.x >> 5;
  const int hlf  = lane >> 4;
  const int colc = lane & 15;
  const int m0 = blockIdx.y * 128 + wv * 32;   // wave owns rows m0..m0+31
  const int n0 = blockIdx.x * 64;

  v8f acc[2][4];
#pragma unroll
  for (int mi = 0; mi < 2; ++mi)
#pragma unroll
    for (int t = 0; t < 4; ++t) acc[mi][t] = vzero8f();

  for (int k0 = 0; k0 < K; k0 += 32) {
    const v16h a0 = load_a_frag(A, K, m0, k0);
    const v16h a1 = load_a_frag(A, K, m0 + 16, k0);
#pragma unroll
    for (int t = 0; t < 4; ++t) {
      const v16h b = load_b_frag(WT, K, n0 + 16 * t, k0);
      acc[0][t] = wmma_f16(a0, b, acc[0][t]);
      acc[1][t] = wmma_f16(a1, b, acc[1][t]);
    }
  }

#pragma unroll
  for (int t = 0; t < 4; ++t) {
    const int n = n0 + 16 * t + colc;
    const float bn = bias[n];
#pragma unroll
    for (int mi = 0; mi < 2; ++mi) {
#pragma unroll
      for (int r = 0; r < 8; ++r) {
        const int m = m0 + mi * 16 + hlf * 8 + r;  // C layout rows
        float c = acc[mi][t][r] + bn;
        if (MODE == 0) {
          const int which = n >> 10;          // 0=q 1=k 2=v
          const int d     = n & 1023;
          const int head  = d >> 6;
          const int hd    = d & 63;
          const int bb    = m >> 11;          // / L
          const int l     = m & 2047;
          const size_t qk = (((size_t)bb * AH + head) * AL + l) * AHD + hd;
          if (which == 0)      qf[qk] = (_Float16)(c * 0.125f);  // 1/sqrt(64)
          else if (which == 1) kf[qk] = (_Float16)c;
          else
            vtp[((((size_t)bb * 2 + (l & 1)) * AH + head) * AHD + hd) * ASUB +
                (l >> 1)] = (_Float16)c;
        } else if (MODE == 1) {
          const size_t idx = (size_t)m * N + n;
          out32[idx] = c + res[idx];
        } else {
          const float gl = 0.5f * c * (1.0f + erff(c * 0.70710678118654752f));
          out16[(size_t)m * N + n] = (_Float16)gl;
        }
      }
    }
  }
}

// ---------------------------------------------------------------------------
// Dilated attention = dense causal sliding-window flash attention per
// (batch, head, parity).  Block: 128 threads (4 waves), 64 queries / block,
// 16 queries / wave.  32-key steps: 4 WMMA for QK^T, 4 WMMA for P*V.
// ---------------------------------------------------------------------------
__global__ void attn_flash(const _Float16* __restrict__ qm,
                           const _Float16* __restrict__ km,
                           const _Float16* __restrict__ vtp,
                           _Float16* __restrict__ om) {
  __shared__ _Float16 pshm[4][16][32];   // per-wave P tile (C->A relayout)

  const int tid  = threadIdx.x;
  const int wv   = tid >> 5;
  const int lane = tid & 31;
  const int hlf  = lane >> 4;
  const int colc = lane & 15;

  const int qb0 = blockIdx.x * 64;       // block query start (sub-index)
  const int par = blockIdx.y;            // parity
  const int bh  = blockIdx.z;            // b*H + h
  const int b   = bh >> 4;
  const int h   = bh & 15;
  const int qs0 = qb0 + wv * 16;         // wave query start

  // q,k: [b,h,l,hd]; sub-index s -> l = par + 2*s  => row stride 2*AHD
  const _Float16* qb = qm + (size_t)bh * AL * AHD + par * AHD;
  const _Float16* kb = km + (size_t)bh * AL * AHD + par * AHD;
  // v: [b, parity, h, hd, s]  (N-major for B fragments)
  const _Float16* vb = vtp + (((size_t)b * 2 + par) * AH + h) * (size_t)AHD * ASUB;

  const v16h qa0 = load_a_frag(qb, 2 * AHD, qs0, 0);
  const v16h qa1 = load_a_frag(qb, 2 * AHD, qs0, 32);

  float mrow[8], lrow[8];
  v8f oacc[4];
#pragma unroll
  for (int r = 0; r < 8; ++r) { mrow[r] = -1e30f; lrow[r] = 0.f; }
#pragma unroll
  for (int t = 0; t < 4; ++t) oacc[t] = vzero8f();

  int kb_lo = qb0 - AWIN;
  if (kb_lo < 0) kb_lo = 0;
  kb_lo &= ~31;
  const int kb_hi = qb0 + 64;            // block-uniform trip count

  for (int kk = kb_lo; kk < kb_hi; kk += 32) {
    // ---- scores: 16 queries x 32 keys, contraction over HD=64 (2 k-steps)
    v8f s0 = vzero8f(), s1 = vzero8f();
    s0 = wmma_f16(qa0, load_b_frag(kb, 2 * AHD, kk, 0), s0);
    s0 = wmma_f16(qa1, load_b_frag(kb, 2 * AHD, kk, 32), s0);
    s1 = wmma_f16(qa0, load_b_frag(kb, 2 * AHD, kk + 16, 0), s1);
    s1 = wmma_f16(qa1, load_b_frag(kb, 2 * AHD, kk + 16, 32), s1);

    float corr[8];
    __syncthreads();                     // pshm reuse fence (uniform bounds)
#pragma unroll
    for (int r = 0; r < 8; ++r) {
      const int qrow = qs0 + hlf * 8 + r;
      const int key0 = kk + colc;
      const int key1 = kk + 16 + colc;
      float v0 = s0[r], v1 = s1[r];
      const bool ok0 = (key0 <= qrow) && (qrow - key0 <= AWIN);
      const bool ok1 = (key1 <= qrow) && (qrow - key1 <= AWIN);
      v0 = ok0 ? v0 : -1e30f;
      v1 = ok1 ? v1 : -1e30f;
      // row max across the 16-lane half
      float mx = fmaxf(v0, v1);
      mx = fmaxf(mx, __shfl_xor(mx, 1, 16));
      mx = fmaxf(mx, __shfl_xor(mx, 2, 16));
      mx = fmaxf(mx, __shfl_xor(mx, 4, 16));
      mx = fmaxf(mx, __shfl_xor(mx, 8, 16));
      const float mnew = fmaxf(mrow[r], mx);
      const bool safe  = mnew > -1e29f;
      const float cr = safe ? __expf(mrow[r] - mnew) : 1.0f;
      const float p0 = safe ? __expf(v0 - mnew) : 0.0f;
      const float p1 = safe ? __expf(v1 - mnew) : 0.0f;
      float ps = p0 + p1;
      ps += __shfl_xor(ps, 1, 16);
      ps += __shfl_xor(ps, 2, 16);
      ps += __shfl_xor(ps, 4, 16);
      ps += __shfl_xor(ps, 8, 16);
      lrow[r] = lrow[r] * cr + ps;
      mrow[r] = mnew;
      corr[r] = cr;
      pshm[wv][hlf * 8 + r][colc]      = (_Float16)p0;
      pshm[wv][hlf * 8 + r][colc + 16] = (_Float16)p1;
    }
    __syncthreads();                     // stores visible before A-frag loads

    // rescale running O
#pragma unroll
    for (int t = 0; t < 4; ++t)
#pragma unroll
      for (int r = 0; r < 8; ++r) oacc[t][r] *= corr[r];

    // P tile as A fragment (16x32, M=query, K=key) from LDS
    v16h pa;
    {
      const _Float16* pp = &pshm[wv][colc][hlf * 16];
#pragma unroll
      for (int i = 0; i < 8; ++i) { pa[i] = pp[i]; pa[i + 8] = pp[8 + i]; }
    }
    // O += P * V  (contraction over 32 keys, 4 feature tiles of 16)
#pragma unroll
    for (int t = 0; t < 4; ++t)
      oacc[t] = wmma_f16(pa, load_b_frag(vb, ASUB, 16 * t, kk), oacc[t]);
  }

  // epilogue: O / l, scatter to o[b, l, h*64 + feat]  (f16 for Wout GEMM)
#pragma unroll
  for (int t = 0; t < 4; ++t) {
#pragma unroll
    for (int r = 0; r < 8; ++r) {
      const int qrow = qs0 + hlf * 8 + r;
      const int labs = par + 2 * qrow;
      const float den = lrow[r];
      const float val = (den > 0.f) ? (oacc[t][r] / den) : 0.f;
      om[((size_t)b * AL + labs) * AD + h * AHD + 16 * t + colc] = (_Float16)val;
    }
  }
}

// ---------------------------------------------------------------------------
extern "C" void kernel_launch(void* const* d_in, const int* in_sizes, int n_in,
                              void* d_out, int out_size, void* d_ws,
                              size_t ws_size, hipStream_t stream) {
  (void)in_sizes; (void)n_in; (void)out_size; (void)ws_size;

  const float* x     = (const float*)d_in[0];
  const float* ln1_g = (const float*)d_in[1];
  const float* ln1_b = (const float*)d_in[2];
  const float* Wqkv  = (const float*)d_in[3];
  const float* bqkv  = (const float*)d_in[4];
  const float* Wout  = (const float*)d_in[5];
  const float* bout  = (const float*)d_in[6];
  const float* ln2_g = (const float*)d_in[7];
  const float* ln2_b = (const float*)d_in[8];
  const float* W1    = (const float*)d_in[9];
  const float* b1    = (const float*)d_in[10];
  const float* W2    = (const float*)d_in[11];
  const float* b2    = (const float*)d_in[12];
  float* out = (float*)d_out;

  const size_t M = (size_t)AB * AL;   // 4096 rows

  // workspace carve-up (256B-aligned)
  char* wsp = (char*)d_ws;
  auto alloc = [&](size_t bytes) {
    char* p = wsp;
    wsp += (bytes + 255) & ~(size_t)255;
    return p;
  };
  _Float16* h1    = (_Float16*)alloc(M * AD * 2);              // LN1 out
  _Float16* wqkvT = (_Float16*)alloc((size_t)3 * AD * AD * 2); // [3D][D]
  _Float16* woutT = (_Float16*)alloc((size_t)AD * AD * 2);     // [D][D]
  _Float16* w1T   = (_Float16*)alloc((size_t)4 * AD * AD * 2); // [4D][D]
  _Float16* w2T   = (_Float16*)alloc((size_t)4 * AD * AD * 2); // [D][4D]
  _Float16* qf    = (_Float16*)alloc((size_t)AB * AH * AL * AHD * 2);
  _Float16* kf    = (_Float16*)alloc((size_t)AB * AH * AL * AHD * 2);
  _Float16* vtp   = (_Float16*)alloc((size_t)AB * 2 * AH * AHD * ASUB * 2);
  _Float16* of    = (_Float16*)alloc(M * AD * 2);              // attn out
  float*    x1    = (float*)   alloc(M * AD * 4);              // residual 1
  _Float16* h2    = (_Float16*)alloc(M * AD * 2);              // LN2 out
  _Float16* ffn   = (_Float16*)alloc(M * (size_t)4 * AD * 2);  // gelu out

  const dim3 tb(32, 8);
  // weight transposes (fp32 [K][N] -> f16 [N][K])
  w_transpose_f16<<<dim3(3 * AD / 32, AD / 32), tb, 0, stream>>>(Wqkv, wqkvT, AD, 3 * AD);
  w_transpose_f16<<<dim3(AD / 32, AD / 32),     tb, 0, stream>>>(Wout, woutT, AD, AD);
  w_transpose_f16<<<dim3(4 * AD / 32, AD / 32), tb, 0, stream>>>(W1,   w1T,   AD, 4 * AD);
  w_transpose_f16<<<dim3(AD / 32, 4 * AD / 32), tb, 0, stream>>>(W2,   w2T,   4 * AD, AD);

  // LN1
  ln_to_f16<<<(int)M, 256, 0, stream>>>(x, ln1_g, ln1_b, h1);

  // QKV projection (M=4096, N=3072, K=1024) with q/k/v scatter epilogue
  gemm_wmma<0><<<dim3(3 * AD / 64, (int)M / 128), 128, 0, stream>>>(
      h1, wqkvT, bqkv, nullptr, nullptr, nullptr, qf, kf, vtp,
      (int)M, 3 * AD, AD);

  // dilated attention (per b,h,parity flash attention)
  attn_flash<<<dim3(ASUB / 64, 2, AB * AH), 128, 0, stream>>>(qf, kf, vtp, of);

  // x1 = x + o @ Wout + bout   (M=4096, N=1024, K=1024)
  gemm_wmma<1><<<dim3(AD / 64, (int)M / 128), 128, 0, stream>>>(
      of, woutT, bout, x, x1, nullptr, nullptr, nullptr, nullptr,
      (int)M, AD, AD);

  // LN2
  ln_to_f16<<<(int)M, 256, 0, stream>>>(x1, ln2_g, ln2_b, h2);

  // f = gelu(h2 @ W1 + b1)   (M=4096, N=4096, K=1024)
  gemm_wmma<2><<<dim3(4 * AD / 64, (int)M / 128), 128, 0, stream>>>(
      h2, w1T, b1, nullptr, nullptr, ffn, nullptr, nullptr, nullptr,
      (int)M, 4 * AD, AD);

  // out = x1 + f @ W2 + b2   (M=4096, N=1024, K=4096)
  gemm_wmma<1><<<dim3(AD / 64, (int)M / 128), 128, 0, stream>>>(
      ffn, w2T, b2, x1, out, nullptr, nullptr, nullptr, nullptr,
      (int)M, AD, 4 * AD);
}
